// HeteroGraph_38757784879708
// MI455X (gfx1250) — matrix operand order repaired and checked by reading, not compiled, not run on hardware
//
#include <hip/hip_runtime.h>
#include <hip/hip_bf16.h>

// ---------------------------------------------------------------------------
// HeteroGraph SAGE GNN for MI455X (gfx1250).
//
// - Side aggregations (agr->sub, urb->sub) are layer-invariant: computed once.
// - Root weights fold: h@Wr_ss + h@Wr_as + h@Wr_us = h@(sum).
// - All activations are kept as bf16 in memory; dense math is
//   v_wmma_f32_16x16x32_bf16 with f32 accumulators.  f32->bf16 conversion
//   happens once per element in bandwidth-bound passes, never in the GEMM.
// - Edge aggregation: bf16 gather + f32 atomicAdd scatter into an L2-resident
//   51MB accumulator (192MB L2 => L2-bound, not HBM-bound).
// - Weights are pre-packed into WMMA B-fragment order so each lane fetches a
//   fragment as two contiguous b128 loads.
// ---------------------------------------------------------------------------

typedef __attribute__((ext_vector_type(16))) __bf16 v16bf;
typedef __attribute__((ext_vector_type(8)))  __bf16 v8bf;
typedef __attribute__((ext_vector_type(8)))  float  v8f;
typedef __attribute__((ext_vector_type(4)))  float  v4f;

#define HH   128   // hidden / sub feature dim
#define DHRU 32    // agr/urb feature dim
#define NOUT 16
#define NLAY 3

__device__ __forceinline__ __bf16 f2bf(float f) {
  unsigned u = __builtin_bit_cast(unsigned, f);
  u += 0x7FFFu + ((u >> 16) & 1u);           // round-to-nearest-even
  unsigned short s = (unsigned short)(u >> 16);
  return __builtin_bit_cast(__bf16, s);
}

__device__ __forceinline__ float bf2f(__bf16 b) {
  unsigned u = (unsigned)__builtin_bit_cast(unsigned short, b) << 16;
  return __builtin_bit_cast(float, u);
}

// 16x32 bf16 A-fragment from a row-major bf16 matrix.
// Contiguous K layout (matches packed B):
//   lanes 0-15 : elems 0..15 -> K=kb+0..15
//   lanes 16-31: elems 0..15 -> K=kb+16..31
// => two contiguous b128 loads per lane.
__device__ __forceinline__ v16bf load_a_bf(const __bf16* __restrict__ X,
                                           int ld, int row, int kb, int hi) {
  const __bf16* p = X + (size_t)row * ld + kb + hi * 16;
  v8bf a = *(const v8bf*)(p);
  v8bf b = *(const v8bf*)(p + 8);
  return __builtin_shufflevector(a, b, 0, 1, 2, 3, 4, 5, 6, 7,
                                 8, 9, 10, 11, 12, 13, 14, 15);
}

__device__ __forceinline__ v8f wmma_bf16(v16bf a, v16bf b, v8f c) {
  return __builtin_amdgcn_wmma_f32_16x16x32_bf16(
      /*neg_a=*/false, a, /*neg_b=*/false, b,
      /*c_mod=*/(short)0, c, /*reuse_a=*/false, /*reuse_b=*/false);
}

// ---------------------------------------------------------------------------
// Utility kernels
// ---------------------------------------------------------------------------
__global__ void fill4_kernel(float4* __restrict__ p, long long n4) {
  long long t = (long long)blockIdx.x * blockDim.x + threadIdx.x;
  if (t < n4) p[t] = make_float4(0.f, 0.f, 0.f, 0.f);
}

__global__ void deg_count_kernel(const int* __restrict__ dst,
                                 float* __restrict__ deg, int e) {
  int t = blockIdx.x * blockDim.x + threadIdx.x;
  if (t < e) atomicAdd(&deg[dst[t]], 1.0f);
}

__global__ void rdeg_kernel(float* __restrict__ deg, int n) {
  int t = blockIdx.x * blockDim.x + threadIdx.x;
  if (t < n) deg[t] = 1.0f / fmaxf(deg[t], 1.0f);
}

// f32 -> bf16 conversion, 8 elems/thread, optional per-row scale
// (row = (element_index / 8) >> rshift, with 8 | row length).
__global__ void cvt_bf16_kernel(const float* __restrict__ src,
                                __bf16* __restrict__ dst,
                                const float* __restrict__ rowscale,
                                long long n8, int rshift) {
  long long t = (long long)blockIdx.x * blockDim.x + threadIdx.x;
  if (t >= n8) return;
  float sc = rowscale ? rowscale[t >> rshift] : 1.0f;
  v4f a = *(const v4f*)(src + t * 8);
  v4f b = *(const v4f*)(src + t * 8 + 4);
  v8bf r;
#pragma unroll
  for (int e = 0; e < 4; ++e) {
    r[e]     = f2bf(a[e] * sc);
    r[4 + e] = f2bf(b[e] * sc);
  }
  *(v8bf*)(dst + t * 8) = r;
}

// f32-source gather/scatter (one float4 of one edge per thread).
__global__ void scatter_add_f32_kernel(const float* __restrict__ x,
                                       const int* __restrict__ src,
                                       const int* __restrict__ dst,
                                       float* __restrict__ out,
                                       long long n_work, int shift, int d) {
  long long t = (long long)blockIdx.x * blockDim.x + threadIdx.x;
  if (t >= n_work) return;
  int e = (int)(t >> shift);
  int c = ((int)t & ((1 << shift) - 1)) * 4;
  int s  = src[e];
  int dn = dst[e];
  const float4 v = *(const float4*)(x + (size_t)s * d + c);
  float* o = out + (size_t)dn * d + c;
  atomicAdd(o + 0, v.x);
  atomicAdd(o + 1, v.y);
  atomicAdd(o + 2, v.z);
  atomicAdd(o + 3, v.w);
}

// bf16-source gather (b128), f32 atomic scatter (8 elems of one edge/thread).
__global__ void scatter_add_bf16_kernel(const __bf16* __restrict__ x,
                                        const int* __restrict__ src,
                                        const int* __restrict__ dst,
                                        float* __restrict__ out,
                                        long long n_work, int shift, int d) {
  long long t = (long long)blockIdx.x * blockDim.x + threadIdx.x;
  if (t >= n_work) return;
  int e = (int)(t >> shift);
  int c = ((int)t & ((1 << shift) - 1)) * 8;
  int s  = src[e];
  int dn = dst[e];
  v8bf v = *(const v8bf*)(x + (size_t)s * d + c);
  float* o = out + (size_t)dn * d + c;
#pragma unroll
  for (int j = 0; j < 8; ++j) atomicAdd(o + j, bf2f(v[j]));
}

// ---------------------------------------------------------------------------
// Weight prep: bf16 conversion into WMMA B-fragment layout.
// Packed layout per (kstep, ntile): 32 lanes x 16 contiguous bf16 (lane l
// holds column n = nt*16 + (l&15), K = ks*32 + (l>>4)*16 + e).
// ---------------------------------------------------------------------------
__global__ void prep_weights_kernel(
    const float* __restrict__ Wl_ss, const float* __restrict__ Wr_ss,
    const float* __restrict__ Wr_as, const float* __restrict__ Wr_us,
    const float* __restrict__ Wl_as, const float* __restrict__ Wl_us,
    const float* __restrict__ bl_ss, const float* __restrict__ bl_as,
    const float* __restrict__ bl_us,
    __bf16* __restrict__ wl_pk, __bf16* __restrict__ wr_pk,
    __bf16* __restrict__ wla_pk, __bf16* __restrict__ wlu_pk,
    float* __restrict__ bias) {
  const int NWW = NLAY * HH * HH;     // wl + wr (K=128)
  const int NWS = NLAY * DHRU * HH;   // wla + wlu (K=32)
  const int NB  = NLAY * HH;
  int tid = blockIdx.x * blockDim.x + threadIdx.x;
  if (tid < NWW) {
    int l = tid / (HH * HH);
    int r = tid % (HH * HH);
    int e    = r & 15;
    int lane = (r >> 4) & 31;
    int nt   = (r >> 9) & 7;
    int ks   = r >> 12;                      // 0..3
    int k = ks * 32 + ((lane >> 4) << 4) + e;
    int n = nt * 16 + (lane & 15);
    size_t w = (size_t)l * HH * HH + (size_t)k * HH + n;
    wl_pk[tid] = f2bf(Wl_ss[w]);
    wr_pk[tid] = f2bf(Wr_ss[w] + Wr_as[w] + Wr_us[w]);
  } else if (tid < NWW + NWS) {
    int t = tid - NWW;
    int l = t / (DHRU * HH);
    int r = t % (DHRU * HH);
    int e    = r & 15;
    int lane = (r >> 4) & 31;
    int nt   = (r >> 9) & 7;
    int k = ((lane >> 4) << 4) + e;          // K=32, single kstep
    int n = nt * 16 + (lane & 15);
    size_t w = (size_t)l * DHRU * HH + (size_t)k * HH + n;
    wla_pk[t] = f2bf(Wl_as[w]);
    wlu_pk[t] = f2bf(Wl_us[w]);
  } else if (tid < NWW + NWS + NB) {
    int t = tid - NWW - NWS;
    bias[t] = bl_ss[t] + bl_as[t] + bl_us[t];
  }
}

// ---------------------------------------------------------------------------
// Fused layer: h_next = relu(aggs@Wl + h@Wr + agg_as@Wla + agg_us@Wlu + bias)
// (aggs already deg-scaled during bf16 conversion for layers > 0).
// Block = 256 threads = 8 waves; wave w owns N-tile w; block owns 32 rows
// (2 M-tiles), so every B-fragment is reused across 2 WMMAs.
// 20 v_wmma_f32_16x16x32_bf16 per wave; inputs/outputs all bf16.
// ---------------------------------------------------------------------------
__global__ __launch_bounds__(256) void layer_kernel(
    const __bf16* __restrict__ hbf, const __bf16* __restrict__ aggbf,
    const __bf16* __restrict__ aggAbf, const __bf16* __restrict__ aggUbf,
    const __bf16* __restrict__ Wl, const __bf16* __restrict__ Wr,
    const __bf16* __restrict__ Wla, const __bf16* __restrict__ Wlu,
    const float* __restrict__ bias, __bf16* __restrict__ h_out) {
  const int lane = threadIdx.x & 31;
  const int wave = threadIdx.x >> 5;       // N-tile 0..7
  const int hi   = lane >> 4;
  const int lo   = lane & 15;
  const int row0 = blockIdx.x * 32;
  const int rA   = row0 + lo;              // M-tile 0 fragment row
  const int rB   = row0 + 16 + lo;         // M-tile 1 fragment row
  const int col  = wave * 16 + lo;         // output column

  const float b = bias[col];
  v8f acc0 = {b, b, b, b, b, b, b, b};
  v8f acc1 = acc0;

#pragma unroll
  for (int ks = 0; ks < HH; ks += 32) {
    const __bf16* wp = Wl + ((size_t)((ks >> 5) * 8 + wave) * 512 + lane * 16);
    v16bf w0 = *(const v16bf*)(wp);
    v16bf a  = load_a_bf(aggbf, HH, rA, ks, hi);
    acc0 = wmma_bf16(a, w0, acc0);
    a    = load_a_bf(aggbf, HH, rB, ks, hi);
    acc1 = wmma_bf16(a, w0, acc1);

    const __bf16* wq = Wr + ((size_t)((ks >> 5) * 8 + wave) * 512 + lane * 16);
    v16bf w1 = *(const v16bf*)(wq);
    a    = load_a_bf(hbf, HH, rA, ks, hi);
    acc0 = wmma_bf16(a, w1, acc0);
    a    = load_a_bf(hbf, HH, rB, ks, hi);
    acc1 = wmma_bf16(a, w1, acc1);
  }
  // K=32 side aggregations (layer-invariant inputs)
  {
    v16bf w = *(const v16bf*)(Wla + ((size_t)wave * 512 + lane * 16));
    v16bf a = load_a_bf(aggAbf, DHRU, rA, 0, hi);
    acc0 = wmma_bf16(a, w, acc0);
    a    = load_a_bf(aggAbf, DHRU, rB, 0, hi);
    acc1 = wmma_bf16(a, w, acc1);

    w = *(const v16bf*)(Wlu + ((size_t)wave * 512 + lane * 16));
    a    = load_a_bf(aggUbf, DHRU, rA, 0, hi);
    acc0 = wmma_bf16(a, w, acc0);
    a    = load_a_bf(aggUbf, DHRU, rB, 0, hi);
    acc1 = wmma_bf16(a, w, acc1);
  }
  // ReLU + bf16 store (D layout: VGPR r -> M = r + hi*8, N = lo)
  __bf16* outp = h_out + (size_t)row0 * HH;
#pragma unroll
  for (int r = 0; r < 8; ++r) {
    float v0 = acc0[r];
    float v1 = acc1[r];
    v0 = v0 > 0.f ? v0 : 0.f;
    v1 = v1 > 0.f ? v1 : 0.f;
    outp[(size_t)(r + hi * 8) * HH + col]      = f2bf(v0);
    outp[(size_t)(16 + r + hi * 8) * HH + col] = f2bf(v1);
  }
}

// ---------------------------------------------------------------------------
// Head: logits = h @ Wf + bf, softmax over 16 classes.  0.4 GFLOP total —
// plain VALU + LDS softmax.  Block = 256 threads = 16 rows x 16 classes.
// ---------------------------------------------------------------------------
__global__ __launch_bounds__(256) void head_kernel(
    const __bf16* __restrict__ h, const float* __restrict__ Wf,
    const float* __restrict__ bf_, float* __restrict__ out) {
  __shared__ float sm[16][17];
  const int cid = threadIdx.x & 15;
  const int rid = threadIdx.x >> 4;
  const int row = blockIdx.x * 16 + rid;
  const __bf16* hr = h + (size_t)row * HH;
  float acc = bf_[cid];
#pragma unroll 4
  for (int k = 0; k < HH; ++k) acc += bf2f(hr[k]) * Wf[k * NOUT + cid];
  sm[rid][cid] = acc;
  __syncthreads();
  float m = -1e30f;
#pragma unroll
  for (int c = 0; c < NOUT; ++c) m = fmaxf(m, sm[rid][c]);
  __syncthreads();
  float e = __expf(acc - m);
  sm[rid][cid] = e;
  __syncthreads();
  float s = 0.f;
#pragma unroll
  for (int c = 0; c < NOUT; ++c) s += sm[rid][c];
  out[(size_t)row * NOUT + cid] = e / s;
}

// ---------------------------------------------------------------------------
// Host-side orchestration
// ---------------------------------------------------------------------------
static inline unsigned nblk(long long n, int t) {
  return (unsigned)((n + t - 1) / t);
}

extern "C" void kernel_launch(void* const* d_in, const int* in_sizes, int n_in,
                              void* d_out, int out_size, void* d_ws,
                              size_t ws_size, hipStream_t stream) {
  (void)n_in; (void)out_size; (void)ws_size;

  const float* x_sub = (const float*)d_in[0];
  const float* x_agr = (const float*)d_in[1];
  const float* x_urb = (const float*)d_in[2];
  const int* ss_src  = (const int*)d_in[3];
  const int* ss_dst  = (const int*)d_in[4];
  const int* as_src  = (const int*)d_in[5];
  const int* as_dst  = (const int*)d_in[6];
  const int* us_src  = (const int*)d_in[7];
  const int* us_dst  = (const int*)d_in[8];
  const float* Wl_ss = (const float*)d_in[9];
  const float* bl_ss = (const float*)d_in[10];
  const float* Wr_ss = (const float*)d_in[11];
  const float* Wl_as = (const float*)d_in[12];
  const float* bl_as = (const float*)d_in[13];
  const float* Wr_as = (const float*)d_in[14];
  const float* Wl_us = (const float*)d_in[15];
  const float* bl_us = (const float*)d_in[16];
  const float* Wr_us = (const float*)d_in[17];
  const float* Wf    = (const float*)d_in[18];
  const float* bfv   = (const float*)d_in[19];

  const int n_sub = in_sizes[0] / HH;
  const int e_ss  = in_sizes[3];
  const int e_as  = in_sizes[5];
  const int e_us  = in_sizes[7];

  // ---- workspace carve (all 256B aligned) ----
  char* w = (char*)d_ws;
  auto carve = [&](size_t bytes) {
    void* p = (void*)w;
    w += (bytes + 255) & ~(size_t)255;
    return p;
  };
  float*  agg    = (float*)carve((size_t)n_sub * HH * 4);     // f32 scatter acc
  float*  aggA   = (float*)carve((size_t)n_sub * DHRU * 4);
  float*  aggU   = (float*)carve((size_t)n_sub * DHRU * 4);
  float*  rdeg   = (float*)carve((size_t)n_sub * 4);
  __bf16* xbf    = (__bf16*)carve((size_t)n_sub * HH * 2);    // bf16 activations
  __bf16* hbfA   = (__bf16*)carve((size_t)n_sub * HH * 2);
  __bf16* hbfB   = (__bf16*)carve((size_t)n_sub * HH * 2);
  __bf16* aggbf  = (__bf16*)carve((size_t)n_sub * HH * 2);
  __bf16* aggAbf = (__bf16*)carve((size_t)n_sub * DHRU * 2);
  __bf16* aggUbf = (__bf16*)carve((size_t)n_sub * DHRU * 2);
  __bf16* wl_pk  = (__bf16*)carve((size_t)NLAY * HH * HH * 2);
  __bf16* wr_pk  = (__bf16*)carve((size_t)NLAY * HH * HH * 2);
  __bf16* wla_pk = (__bf16*)carve((size_t)NLAY * DHRU * HH * 2);
  __bf16* wlu_pk = (__bf16*)carve((size_t)NLAY * DHRU * HH * 2);
  float*  bias   = (float*)carve((size_t)NLAY * HH * 4);

  // ---- static prep ----
  fill4_kernel<<<nblk((long long)n_sub * DHRU / 4, 256), 256, 0, stream>>>(
      (float4*)aggA, (long long)n_sub * DHRU / 4);
  fill4_kernel<<<nblk((long long)n_sub * DHRU / 4, 256), 256, 0, stream>>>(
      (float4*)aggU, (long long)n_sub * DHRU / 4);
  fill4_kernel<<<nblk((long long)n_sub / 4, 256), 256, 0, stream>>>(
      (float4*)rdeg, (long long)n_sub / 4);

  const int nprep = NLAY * HH * HH + NLAY * DHRU * HH + NLAY * HH;
  prep_weights_kernel<<<nblk(nprep, 256), 256, 0, stream>>>(
      Wl_ss, Wr_ss, Wr_as, Wr_us, Wl_as, Wl_us, bl_ss, bl_as, bl_us,
      wl_pk, wr_pk, wla_pk, wlu_pk, bias);

  deg_count_kernel<<<nblk(e_ss, 256), 256, 0, stream>>>(ss_dst, rdeg, e_ss);
  rdeg_kernel<<<nblk(n_sub, 256), 256, 0, stream>>>(rdeg, n_sub);

  scatter_add_f32_kernel<<<nblk((long long)e_as * (DHRU / 4), 256), 256, 0,
                           stream>>>(x_agr, as_src, as_dst, aggA,
                                     (long long)e_as * (DHRU / 4), 3, DHRU);
  scatter_add_f32_kernel<<<nblk((long long)e_us * (DHRU / 4), 256), 256, 0,
                           stream>>>(x_urb, us_src, us_dst, aggU,
                                     (long long)e_us * (DHRU / 4), 3, DHRU);

  // bf16 copies of layer-invariant activations
  cvt_bf16_kernel<<<nblk((long long)n_sub * HH / 8, 256), 256, 0, stream>>>(
      x_sub, xbf, nullptr, (long long)n_sub * HH / 8, 0);
  cvt_bf16_kernel<<<nblk((long long)n_sub * DHRU / 8, 256), 256, 0, stream>>>(
      aggA, aggAbf, nullptr, (long long)n_sub * DHRU / 8, 0);
  cvt_bf16_kernel<<<nblk((long long)n_sub * DHRU / 8, 256), 256, 0, stream>>>(
      aggU, aggUbf, nullptr, (long long)n_sub * DHRU / 8, 0);

  // ---- 3 layers ----
  const __bf16* hcur = xbf;
  __bf16* bufs[2] = {hbfA, hbfB};
  for (int i = 0; i < NLAY; ++i) {
    fill4_kernel<<<nblk((long long)n_sub * HH / 4, 256), 256, 0, stream>>>(
        (float4*)agg, (long long)n_sub * HH / 4);
    scatter_add_bf16_kernel<<<nblk((long long)e_ss * (HH / 8), 256), 256, 0,
                              stream>>>(hcur, ss_src, ss_dst, agg,
                                        (long long)e_ss * (HH / 8), 4, HH);
    // f32 agg -> bf16, folding the mean (1/deg) for layers > 0
    cvt_bf16_kernel<<<nblk((long long)n_sub * HH / 8, 256), 256, 0, stream>>>(
        agg, aggbf, (i > 0) ? rdeg : nullptr, (long long)n_sub * HH / 8, 4);

    __bf16* hnext = bufs[i & 1];
    layer_kernel<<<n_sub / 32, 256, 0, stream>>>(
        hcur, aggbf, aggAbf, aggUbf,
        wl_pk + (size_t)i * HH * HH, wr_pk + (size_t)i * HH * HH,
        wla_pk + (size_t)i * DHRU * HH, wlu_pk + (size_t)i * DHRU * HH,
        bias + (size_t)i * HH, hnext);
    hcur = hnext;
  }

  // ---- head ----
  head_kernel<<<n_sub / 16, 256, 0, stream>>>(hcur, Wf, bfv, (float*)d_out);
}
